// BertSelfAttention_11003706212619
// MI455X (gfx1250) — matrix operand order repaired and checked by reading, not compiled, not run on hardware
//
#include <hip/hip_runtime.h>
#include <hip/hip_bf16.h>

// ---------------------------------------------------------------------------
// BERT self-attention (post-norm, shared-V) for gfx1250 (MI455X).
// All GEMMs run through v_wmma_f32_16x16x32_f16 (wave32 WMMA).
// Problem sizes: bs=4, N=1024 (32x32), d=256, HEADS=8.
// ---------------------------------------------------------------------------

typedef __attribute__((ext_vector_type(16))) _Float16 v16h;
typedef __attribute__((ext_vector_type(8)))  _Float16 v8h;
typedef __attribute__((ext_vector_type(8)))  float    v8f;

#define HEADS   8
#define NTOK    1024          // 32*32 tokens per batch
#define DMODEL  256
#define BATCH   4
#define LN_EPS  1e-5f

static __device__ inline v16h cat16(v8h lo, v8h hi) {
    return __builtin_shufflevector(lo, hi,
        0, 1, 2, 3, 4, 5, 6, 7, 8, 9, 10, 11, 12, 13, 14, 15);
}

// ---------------------------------------------------------------------------
// WMMA GEMM:  C[M x N] = alpha * A[M x K] * B[N x K]^T
// A, B row-major f16 with contiguous K. C row-major (OutT = float or f16).
// Grid: x = N/128, y = M/128, z = batch.  Block: 256 threads (8 waves, 4x2).
// Each wave computes a 32x64 patch of C (2x4 WMMA accumulators) for higher
// FLOP/byte at the memory pipe (~21 vs 16 for 32x32).
// Batch offsets decompose z = zhi*HEADS + zlo with independent strides.
// ---------------------------------------------------------------------------
template <typename OutT>
__global__ __launch_bounds__(256)
void wmma_gemm_f16(const _Float16* __restrict__ A, const _Float16* __restrict__ B,
                   OutT* __restrict__ C,
                   int lda, int ldb, int ldc, int K, float alpha,
                   long long sAhi, long long sAlo,
                   long long sBhi, long long sBlo,
                   long long sChi, long long sClo)
{
    const int z   = blockIdx.z;
    const int zhi = z / HEADS;
    const int zlo = z % HEADS;
    A += (long long)zhi * sAhi + (long long)zlo * sAlo;
    B += (long long)zhi * sBhi + (long long)zlo * sBlo;
    C += (long long)zhi * sChi + (long long)zlo * sClo;

    const int wid  = threadIdx.x >> 5;   // wave id 0..7
    const int lane = threadIdx.x & 31;
    const int g    = lane >> 4;          // K-group (0/1) per ISA layout
    const int mr   = lane & 15;          // row (A) / col (B,C) within 16

    const int rowBase = blockIdx.y * 128 + (wid & 3) * 32;   // 2 x 16 rows
    const int colBase = blockIdx.x * 128 + (wid >> 2) * 64;  // 4 x 16 cols

    v8f acc[2][4] = {};

    const _Float16* aP[2];
    const _Float16* bP[4];
    #pragma unroll
    for (int i = 0; i < 2; ++i)
        aP[i] = A + (long long)(rowBase + 16 * i + mr) * lda;
    #pragma unroll
    for (int j = 0; j < 4; ++j)
        bP[j] = B + (long long)(colBase + 16 * j + mr) * ldb;

    for (int k0 = 0; k0 < K; k0 += 32) {
        // A operand (16x32): lane-group g holds K = {8g..8g+7} and {16+8g..23+8g}
        v16h a[2];
        #pragma unroll
        for (int i = 0; i < 2; ++i)
            a[i] = cat16(*(const v8h*)(aP[i] + k0 + g * 8),
                         *(const v8h*)(aP[i] + k0 + 16 + g * 8));
        // B operand (32x16): lane-group g holds K = {8g..8g+7} and {16+8g..23+8g}
        v16h b[4];
        #pragma unroll
        for (int j = 0; j < 4; ++j)
            b[j] = cat16(*(const v8h*)(bP[j] + k0 + g * 8),
                         *(const v8h*)(bP[j] + k0 + 16 + g * 8));

        #pragma unroll
        for (int i = 0; i < 2; ++i)
            #pragma unroll
            for (int j = 0; j < 4; ++j)
                acc[i][j] = __builtin_amdgcn_wmma_f32_16x16x32_f16(
                    false, a[i], false, b[j], (short)0, acc[i][j], false, false);
    }

    // C/D layout: VGPR r, lane-group g -> row = 8g + r, col = lane%16
    #pragma unroll
    for (int i = 0; i < 2; ++i) {
        #pragma unroll
        for (int r = 0; r < 8; ++r) {
            const long long row = rowBase + 16 * i + g * 8 + r;
            #pragma unroll
            for (int j = 0; j < 4; ++j)
                C[row * ldc + colBase + 16 * j + mr] = (OutT)(alpha * acc[i][j][r]);
        }
    }
}

// ---------------------------------------------------------------------------
// fp32 -> f16 elementwise conversion (grid-stride)
// ---------------------------------------------------------------------------
__global__ __launch_bounds__(256)
void f32_to_f16(const float* __restrict__ in, _Float16* __restrict__ out, int n)
{
    int i = blockIdx.x * blockDim.x + threadIdx.x;
    const int stride = gridDim.x * blockDim.x;
    for (; i < n; i += stride) out[i] = (_Float16)in[i];
}

// x (b, N, d) fp32 -> xT (b, d, N) f16   (B operand of the ctx GEMM)
__global__ __launch_bounds__(256)
void transpose_x(const float* __restrict__ x, _Float16* __restrict__ xT)
{
    const int i = blockIdx.x * blockDim.x + threadIdx.x;  // < 4*256*1024
    const int b = i >> 18;
    const int r = i & 0x3FFFF;
    const int dcol = r >> 10;
    const int n = r & 1023;
    xT[i] = (_Float16)x[(((long long)b << 10) + n) * DMODEL + dcol];
}

// ---------------------------------------------------------------------------
// In-place softmax over rows of length N=1024 stored in d_out (layout b,n,h,m),
// plus f16 copy in (b,h,n,m) layout for the ctx GEMM. One wave per row.
// ---------------------------------------------------------------------------
__global__ __launch_bounds__(256)
void softmax_rows(float* __restrict__ attnF, _Float16* __restrict__ attnH)
{
    const int row  = blockIdx.x * 8 + (threadIdx.x >> 5);  // (b*H + h)*N + n
    const int lane = threadIdx.x & 31;
    const int b = row >> 13;
    const int h = (row >> 10) & (HEADS - 1);
    const int n = row & (NTOK - 1);

    float* fRow = attnF + ((((long long)b * NTOK + n) * HEADS + h) << 10);
    _Float16* hRow = attnH + ((long long)row << 10);

    float v[32];
    float mx = -3.4e38f;
    #pragma unroll
    for (int i = 0; i < 32; ++i) {
        v[i] = fRow[lane + 32 * i];
        mx = fmaxf(mx, v[i]);
    }
    #pragma unroll
    for (int off = 16; off; off >>= 1) mx = fmaxf(mx, __shfl_xor(mx, off, 32));

    float s = 0.f;
    #pragma unroll
    for (int i = 0; i < 32; ++i) {
        v[i] = __expf(v[i] - mx);
        s += v[i];
    }
    #pragma unroll
    for (int off = 16; off; off >>= 1) s += __shfl_xor(s, off, 32);

    const float inv = 1.f / s;
    #pragma unroll
    for (int i = 0; i < 32; ++i) {
        const float p = v[i] * inv;
        fRow[lane + 32 * i] = p;
        hRow[lane + 32 * i] = (_Float16)p;
    }
}

// ---------------------------------------------------------------------------
// out = LayerNorm(logits + x) * gamma + beta, one 256-thread block per row
// ---------------------------------------------------------------------------
__global__ __launch_bounds__(256)
void add_ln(const float* __restrict__ logits, const float* __restrict__ x,
            const float* __restrict__ gamma, const float* __restrict__ beta,
            float* __restrict__ out)
{
    __shared__ float red[256];
    const int row = blockIdx.x;
    const int t   = threadIdx.x;
    const long long base = (long long)row * DMODEL + t;

    const float v = logits[base] + x[base];
    red[t] = v;
    __syncthreads();
    #pragma unroll
    for (int s = 128; s; s >>= 1) { if (t < s) red[t] += red[t + s]; __syncthreads(); }
    const float mu = red[0] * (1.f / DMODEL);
    __syncthreads();

    const float d = v - mu;
    red[t] = d * d;
    __syncthreads();
    #pragma unroll
    for (int s = 128; s; s >>= 1) { if (t < s) red[t] += red[t + s]; __syncthreads(); }
    const float var = red[0] * (1.f / DMODEL);

    out[base] = d * rsqrtf(var + LN_EPS) * gamma[t] + beta[t];
}

// ---------------------------------------------------------------------------
extern "C" void kernel_launch(void* const* d_in, const int* in_sizes, int n_in,
                              void* d_out, int out_size, void* d_ws, size_t ws_size,
                              hipStream_t stream)
{
    (void)in_sizes; (void)n_in; (void)out_size; (void)ws_size;

    const float* x     = (const float*)d_in[0];  // (4, 32, 32, 256)
    const float* Wq    = (const float*)d_in[1];  // (2048, 256)
    const float* Wk    = (const float*)d_in[2];  // (2048, 256)
    const float* Wv    = (const float*)d_in[3];  // (256, 2048)
    const float* gamma = (const float*)d_in[4];
    const float* beta  = (const float*)d_in[5];

    float* out   = (float*)d_out;                               // 4*1024*256
    float* attnF = out + (long long)BATCH * NTOK * DMODEL;      // 4*8*1024*1024 (b,n,h,m)

    // Workspace carve-out (256B aligned)
    char* ws = (char*)d_ws;
    size_t off = 0;
    auto carve = [&](size_t bytes) {
        void* p = ws + off;
        off += (bytes + 255) & ~(size_t)255;
        return p;
    };
    _Float16* xh  = (_Float16*)carve((size_t)BATCH * NTOK * DMODEL * 2);   // (4096, 256)
    _Float16* xT  = (_Float16*)carve((size_t)BATCH * DMODEL * NTOK * 2);   // (4, 256, 1024)
    _Float16* wqh = (_Float16*)carve((size_t)HEADS * DMODEL * DMODEL * 2); // (2048, 256)
    _Float16* wkh = (_Float16*)carve((size_t)HEADS * DMODEL * DMODEL * 2);
    _Float16* wvh = (_Float16*)carve((size_t)DMODEL * HEADS * DMODEL * 2); // (256, 2048)
    _Float16* Qh  = (_Float16*)carve((size_t)BATCH * NTOK * HEADS * DMODEL * 2); // (4096, 2048)
    _Float16* Kh  = (_Float16*)carve((size_t)BATCH * NTOK * HEADS * DMODEL * 2);
    _Float16* Ph  = (_Float16*)carve((size_t)BATCH * HEADS * NTOK * NTOK * 2);   // attn f16 (b,h,n,m)
    _Float16* ctx = (_Float16*)carve((size_t)BATCH * NTOK * HEADS * DMODEL * 2); // (4096, 2048)
    float*    lg  = (float*)carve((size_t)BATCH * NTOK * DMODEL * 4);            // (4096, 256)

    const dim3 blk(256);

    // f16 conversions + x transpose
    f32_to_f16<<<1024, blk, 0, stream>>>(x,  xh,  BATCH * NTOK * DMODEL);
    f32_to_f16<<<512,  blk, 0, stream>>>(Wq, wqh, HEADS * DMODEL * DMODEL);
    f32_to_f16<<<512,  blk, 0, stream>>>(Wk, wkh, HEADS * DMODEL * DMODEL);
    f32_to_f16<<<512,  blk, 0, stream>>>(Wv, wvh, DMODEL * HEADS * DMODEL);
    transpose_x<<<4096, blk, 0, stream>>>(x, xT);

    // Q = x @ Wq^T   (M=4096, N=2048, K=256)
    wmma_gemm_f16<_Float16><<<dim3(16, 32, 1), blk, 0, stream>>>(
        xh, wqh, Qh, DMODEL, DMODEL, HEADS * DMODEL, DMODEL, 1.f,
        0, 0, 0, 0, 0, 0);
    // K = x @ Wk^T
    wmma_gemm_f16<_Float16><<<dim3(16, 32, 1), blk, 0, stream>>>(
        xh, wkh, Kh, DMODEL, DMODEL, HEADS * DMODEL, DMODEL, 1.f,
        0, 0, 0, 0, 0, 0);

    // scores[b,h] = (Q_bh @ K_bh^T) / 16, written straight into d_out in the
    // permuted (b,n,h,m) layout: ldc = H*N, per-head column-block offset h*N.
    wmma_gemm_f16<float><<<dim3(8, 8, BATCH * HEADS), blk, 0, stream>>>(
        Qh, Kh, attnF,
        HEADS * DMODEL, HEADS * DMODEL, HEADS * NTOK, DMODEL, 0.0625f,
        (long long)NTOK * HEADS * DMODEL, DMODEL,            // A: b, h
        (long long)NTOK * HEADS * DMODEL, DMODEL,            // B: b, h
        (long long)NTOK * HEADS * NTOK,   NTOK);             // C: b, h

    // softmax in-place in d_out + f16 copy for ctx GEMM
    softmax_rows<<<BATCH * HEADS * NTOK / 8, blk, 0, stream>>>(attnF, Ph);

    // ctx[b,h] = attn_bh @ x_b   (A = attn f16 (1024x1024), B = xT_b (256x1024))
    wmma_gemm_f16<_Float16><<<dim3(2, 8, BATCH * HEADS), blk, 0, stream>>>(
        Ph, xT, ctx, NTOK, NTOK, HEADS * DMODEL, NTOK, 1.f,
        (long long)HEADS * NTOK * NTOK, (long long)NTOK * NTOK,  // A: b, h
        (long long)DMODEL * NTOK,       0,                       // B: b only
        (long long)NTOK * HEADS * DMODEL, DMODEL);               // C: b, h

    // logits = ctx_flat @ Wv^T   (M=4096, N=256, K=2048)
    wmma_gemm_f16<float><<<dim3(2, 32, 1), blk, 0, stream>>>(
        ctx, wvh, lg, HEADS * DMODEL, HEADS * DMODEL, DMODEL,
        HEADS * DMODEL, 1.f, 0, 0, 0, 0, 0, 0);

    // out = LayerNorm(logits + x)
    add_ln<<<BATCH * NTOK, blk, 0, stream>>>(lg, x, gamma, beta, out);
}
